// DeformModelEnhanced_970662608901
// MI455X (gfx1250) — matrix operand order repaired and checked by reading, not compiled, not run on hardware
//
#include <hip/hip_runtime.h>
#include <hip/hip_bf16.h>

// ---------------------------------------------------------------------------
// MI455X (gfx1250) implementation.
// Dense math: v_wmma_f32_16x16x32_f16 (f16 A/B, f32 accumulate).
// GEMM B-operands are staged into LDS by the Tensor Data Mover
// (tensor_load_to_lds + s_wait_tensorcnt, double-buffered), read back with
// ds_load_b128; A fragments stream from global as two b128 loads per lane.
// ---------------------------------------------------------------------------

typedef _Float16 f16;
typedef __attribute__((ext_vector_type(16))) _Float16 v16h;
typedef __attribute__((ext_vector_type(8)))  _Float16 v8h;
typedef __attribute__((ext_vector_type(8)))  float    v8f;
typedef __attribute__((ext_vector_type(4)))  unsigned int u32x4;
typedef __attribute__((ext_vector_type(8)))  int i32x8;
typedef __attribute__((ext_vector_type(4)))  int i32x4;

union Frag { v16h v; v8h h[2]; };

#define BBATCH 16
#define KGR    4096
#define NEDGE  65536
#define NEL    69632              // NEDGE + KGR (self loops)
#define NROWS  65536L             // BBATCH * KGR node rows

#if __has_builtin(__builtin_amdgcn_tensor_load_to_lds) && __has_builtin(__builtin_amdgcn_s_wait_tensorcnt)
#define HAVE_TDM 1
#else
#define HAVE_TDM 0
#endif

__device__ __forceinline__ void atomicMaxF(float* a, float v) {
  if (v >= 0.f) atomicMax((int*)a, __float_as_int(v));
  else          atomicMin((unsigned int*)a, __float_as_uint(v));
}

#if HAVE_TDM
// TDM: copy a 2-D tile (4 rows x 1KB, row stride KT*1KB) of the pre-swizzled
// B-fragment slab into LDS at ldsOff. D# per CDNA5 ISA 8.3-8.5:
//   group0: count=1, lds_addr, global_addr[56:0], type=2
//   group1: data_size=3 (8B), tensor_dim0=KT*128, tensor_dim1=4,
//           tile_dim0=128, tile_dim1=4, tensor_dim0_stride=KT*128
__device__ __forceinline__ void tdm_load_b_slice(const f16* gsrc, unsigned ldsOff, int kt128)
{
  unsigned long long ga = (unsigned long long)(size_t)gsrc;
  u32x4 g0 = { 1u, ldsOff, (unsigned)ga,
               (unsigned)((ga >> 32) & 0x01FFFFFFu) | 0x80000000u };
  i32x8 g1 = { (int)0x30000,                      // data_size=3 (8 bytes)
               (int)((kt128 & 0xFFFF) << 16),     // tensor_dim0 lo16
               (int)((kt128 >> 16) | (4 << 16)),  // tensor_dim0 hi / tensor_dim1=4
               (int)(128 << 16),                  // tile_dim0 = 128 elems
               4,                                 // tile_dim1 = 4 rows
               kt128,                             // tensor_dim0_stride lo32
               0, 0 };
  i32x4 z4 = { 0, 0, 0, 0 };
#if __clang_major__ >= 23
  i32x8 z8 = { 0, 0, 0, 0, 0, 0, 0, 0 };
  __builtin_amdgcn_tensor_load_to_lds(g0, g1, z4, z4, z8, 0);
#else
  __builtin_amdgcn_tensor_load_to_lds(g0, g1, z4, z4, 0);
#endif
}
#endif

// ---------------------------------------------------------------------------
// Generic WMMA GEMM:  C[M,N] = A[M,K](f16,row-major) x Bfrag(K,N swizzled)
// Wave tile 16(M) x 64(N); 4 waves/block -> 64x64 per block; all 4 waves share
// the B slab, staged per-K-slice into LDS double buffers by the TDM.
// A-fragment per-lane layout (16-bit A 16x32, ISA 7.12.2):
//   lane = 16*h + row ; halfs 0..7  -> K = kb+8h..kb+8h+7
//                       halfs 8..15 -> K = kb+16+8h..kb+16+8h+7
// B-fragment (pre-swizzled): element j of lane L (tile kt,nt)
//   = W[kt*32 + 16*(j>>3) + 8*(L>>4) + (j&7)][nt*16 + (L&15)]
// C/D layout: VGPR r of lane L -> M = 8*(L>>4)+r, N = L&15.
// ---------------------------------------------------------------------------
__global__ void __launch_bounds__(128)
gemm_wmma_kernel(const f16* __restrict__ A, const f16* __restrict__ Bf,
                 const float* __restrict__ bias,
                 float* __restrict__ C32, f16* __restrict__ C16,
                 int M, int N, int K, int addC, int relu)
{
#if HAVE_TDM
  extern __shared__ char smem[];                // 2 x 4KB B-slice buffers
#endif
  int wave = threadIdx.x >> 5, lane = threadIdx.x & 31;
  int h = lane >> 4, l = lane & 15;
  int mbase = (blockIdx.y * 4 + wave) << 4;
  int nbase = blockIdx.x << 6;
  int KT = K >> 5;
  int nt0 = nbase >> 4;
  bool active = (mbase < M);                    // wave-uniform
  v8f acc[4] = {};
  const f16* arow = active ? (A + (long)(mbase + l) * K + 8 * h) : A;

#if HAVE_TDM
  if (wave == 0) {
    tdm_load_b_slice(Bf + ((long)nt0 * KT << 9), 0u, KT << 7);
    __builtin_amdgcn_s_wait_tensorcnt((short)0);
  }
  __syncthreads();
  int cur = 0;
#endif
  for (int kt = 0; kt < KT; ++kt) {
    Frag a = {};
    if (active) {
      if (kt + 1 < KT) __builtin_prefetch(arow + 32, 0, 1);
      a.h[0] = *(const v8h*)(arow);
      a.h[1] = *(const v8h*)(arow + 16);
      arow += 32;
    }
#if HAVE_TDM
    if (wave == 0 && kt + 1 < KT)
      tdm_load_b_slice(Bf + (((long)nt0 * KT + kt + 1) << 9),
                       (unsigned)(4096 * (cur ^ 1)), KT << 7);
    const f16* bbase = (const f16*)(smem + 4096 * cur) + (lane << 4);
#pragma unroll
    for (int t = 0; t < 4; ++t) {
      const f16* bp = bbase + (t << 9);
      Frag b;
      b.h[0] = *(const v8h*)(bp);
      b.h[1] = *(const v8h*)(bp + 8);
      acc[t] = __builtin_amdgcn_wmma_f32_16x16x32_f16(
          false, a.v, false, b.v, (short)0, acc[t], false, false);
    }
    if (wave == 0 && kt + 1 < KT) __builtin_amdgcn_s_wait_tensorcnt((short)0);
    __syncthreads();
    cur ^= 1;
#else
#pragma unroll
    for (int t = 0; t < 4; ++t) {
      int nt = nt0 + t;
      const f16* bp = Bf + (((long)(nt * KT + kt) * 32 + lane) << 4);
      Frag b;
      b.h[0] = *(const v8h*)(bp);
      b.h[1] = *(const v8h*)(bp + 8);
      acc[t] = __builtin_amdgcn_wmma_f32_16x16x32_f16(
          false, a.v, false, b.v, (short)0, acc[t], false, false);
    }
#endif
  }
  if (!active) return;
#pragma unroll
  for (int t = 0; t < 4; ++t) {
    int n = nbase + (t << 4) + l;
    float bv = bias ? bias[n] : 0.f;
#pragma unroll
    for (int r = 0; r < 8; ++r) {
      int m = mbase + (h << 3) + r;
      long idx = (long)m * N + n;
      float v = acc[t][r] + bv;
      if (addC) v += C32[idx];
      if (relu) v = fmaxf(v, 0.f);
      if (C32) C32[idx] = v;
      if (C16) C16[idx] = (f16)v;
    }
  }
}

// ---------------------------------------------------------------------------
// Implicit-GEMM 3x3 conv, NHWC [16,64,64,128] f16, K = 9 taps * 128 ic = 1152.
// Fused folded-BN (scale/shift), optional residual add, ReLU.
// ---------------------------------------------------------------------------
__global__ void __launch_bounds__(128)
conv3x3_wmma_kernel(const f16* __restrict__ X, const f16* __restrict__ Wf,
                    const float* __restrict__ scale, const float* __restrict__ shift,
                    const f16* __restrict__ resid, f16* __restrict__ Y)
{
  const int KT = 36;                                  // 1152/32
  int wave = threadIdx.x >> 5, lane = threadIdx.x & 31;
  int h = lane >> 4, l = lane & 15;
  int mbase = (blockIdx.y * 4 + wave) << 4;
  int nbase = blockIdx.x << 6;
  int p = mbase + l;
  int b  = p >> 12;
  int py = (p >> 6) & 63;
  int px = p & 63;
  v8f acc[4] = {};
  for (int kt = 0; kt < KT; ++kt) {
    int kb = kt << 5;
    int tap = kb >> 7;                                // K order = (tap, ic)
    int dy = tap / 3 - 1, dx = tap % 3 - 1;
    int iy = py + dy, ix = px + dx;
    Frag a = {};
    if ((unsigned)iy < 64u && (unsigned)ix < 64u) {
      const f16* src = X + ((((long)b * 64 + iy) * 64 + ix) << 7) + (kb & 127) + 8 * h;
      a.h[0] = *(const v8h*)(src);
      a.h[1] = *(const v8h*)(src + 16);
    }
#pragma unroll
    for (int t = 0; t < 4; ++t) {
      int nt = (nbase >> 4) + t;
      const f16* bp = Wf + (((long)(nt * KT + kt) * 32 + lane) << 4);
      Frag bb;
      bb.h[0] = *(const v8h*)(bp);
      bb.h[1] = *(const v8h*)(bp + 8);
      acc[t] = __builtin_amdgcn_wmma_f32_16x16x32_f16(
          false, a.v, false, bb.v, (short)0, acc[t], false, false);
    }
  }
#pragma unroll
  for (int t = 0; t < 4; ++t) {
    int n = nbase + (t << 4) + l;
    float sc = scale[n], sh = shift[n];
#pragma unroll
    for (int r = 0; r < 8; ++r) {
      int m = mbase + (h << 3) + r;
      long idx = ((long)m << 7) + n;
      float v = acc[t][r] * sc + sh;
      if (resid) v += (float)resid[idx];
      Y[idx] = (f16)fmaxf(v, 0.f);
    }
  }
}

// --------------------------- weight preparation ----------------------------
__global__ void swizzle_w_kernel(const float* __restrict__ W, f16* __restrict__ Bf,
                                 int Kpad, int Ksrc, int N)
{
  int idx = blockIdx.x * blockDim.x + threadIdx.x;
  if (idx >= Kpad * N) return;
  int j = idx & 15, lane = (idx >> 4) & 31, tile = idx >> 9;
  int KT = Kpad >> 5;
  int kt = tile % KT, nt = tile / KT;
  int k = (kt << 5) + ((j >> 3) << 4) + ((lane >> 4) << 3) + (j & 7);
  int n = (nt << 4) + (lane & 15);
  Bf[idx] = (f16)((k < Ksrc) ? W[(long)k * N + n] : 0.f);
}

__global__ void swizzle_conv_kernel(const float* __restrict__ w, f16* __restrict__ Bf)
{
  // w: [oc=128][ic=128][3][3] f32 ; K index = tap*128 + ic, N = oc
  int idx = blockIdx.x * blockDim.x + threadIdx.x;
  if (idx >= 1152 * 128) return;
  int j = idx & 15, lane = (idx >> 4) & 31, tile = idx >> 9;
  const int KT = 36;
  int kt = tile % KT, nt = tile / KT;
  int k = (kt << 5) + ((j >> 3) << 4) + ((lane >> 4) << 3) + (j & 7);
  int oc = (nt << 4) + (lane & 15);
  int tap = k >> 7, ic = k & 127;
  Bf[idx] = (f16)w[oc * 1152 + ic * 9 + tap];
}

__global__ void bnfold_kernel(const float* g, const float* b, const float* m,
                              const float* v, float* scale, float* shift)
{
  int c = threadIdx.x;
  float sc = g[c] * rsqrtf(v[c] + 1e-5f);
  scale[c] = sc;
  shift[c] = b[c] - m[c] * sc;
}

// ------------------------------- CNN glue ----------------------------------
__global__ void stem_kernel(const float* __restrict__ sensor, const float* __restrict__ w,
                            const float* __restrict__ scale, const float* __restrict__ shift,
                            f16* __restrict__ out)
{
  int idx = blockIdx.x * blockDim.x + threadIdx.x;
  if (idx >= 16 * 64 * 64 * 128) return;
  int oc = idx & 127;
  int p = idx >> 7;
  int ox = p & 63, oy = (p >> 6) & 63, b = p >> 12;
  const float* img = sensor + (long)b * 128 * 128;
  const float* wk = w + oc * 49;
  float mx = 0.f;                                // relu floor
  for (int py = 0; py < 2; ++py)
    for (int pxx = 0; pxx < 2; ++pxx) {
      int cy = 2 * oy + py, cx = 2 * ox + pxx;
      float s = 0.f;
      for (int ky = 0; ky < 7; ++ky) {
        int iy = cy + ky - 3;
        if ((unsigned)iy >= 128u) continue;
        for (int kx = 0; kx < 7; ++kx) {
          int ix = cx + kx - 3;
          if ((unsigned)ix >= 128u) continue;
          s += img[iy * 128 + ix] * wk[ky * 7 + kx];
        }
      }
      mx = fmaxf(mx, fmaxf(s * scale[oc] + shift[oc], 0.f));
    }
  out[idx] = (f16)mx;
}

__global__ void avgpool_kernel(const f16* __restrict__ X, float* __restrict__ out)
{
  int b = blockIdx.x >> 7, c = blockIdx.x & 127;
  float s = 0.f;
  for (int p = threadIdx.x; p < 4096; p += 256)
    s += (float)X[(((long)b * 4096 + p) << 7) + c];
  __shared__ float sm[256];
  sm[threadIdx.x] = s; __syncthreads();
  for (int st = 128; st > 0; st >>= 1) {
    if (threadIdx.x < st) sm[threadIdx.x] += sm[threadIdx.x + st];
    __syncthreads();
  }
  if (threadIdx.x == 0) out[blockIdx.x] = sm[0] * (1.f / 4096.f);
}

__global__ void head512_kernel(const float* __restrict__ pool, const float* __restrict__ w,
                               const float* __restrict__ bias, float* __restrict__ feat)
{
  int idx = blockIdx.x * blockDim.x + threadIdx.x;
  if (idx >= 16 * 512) return;
  int b = idx >> 9, o = idx & 511;
  float s = bias[o];
  for (int c = 0; c < 128; ++c) s += pool[b * 128 + c] * w[c * 512 + o];
  feat[idx] = fmaxf(s, 0.f);
}

__global__ void xcat_kernel(const float* __restrict__ feat, const float* __restrict__ tn,
                            f16* __restrict__ xcat)
{
  int idx = blockIdx.x * blockDim.x + threadIdx.x;
  if (idx >= 16 * 544) return;
  int b = idx / 544, c = idx % 544;
  float v = 0.f;
  if (c < 512) v = feat[b * 512 + c];
  else if (c < 525) {
    int j = c - 512;
    float t = tn[b];
    if (j == 0) v = t;
    else {
      float f = (float)(1 << ((j - 1) >> 1));
      v = (j & 1) ? sinf(t * f) : cosf(t * f);
    }
  }
  xcat[idx] = (f16)v;
}

__global__ void fco_kernel(const float* __restrict__ coords, f16* __restrict__ fco)
{
  int idx = blockIdx.x * blockDim.x + threadIdx.x;
  if (idx >= 4096 * 64) return;
  int k = idx >> 6, c = idx & 63;
  float v = 0.f;
  if (c < 3) v = coords[k * 3 + c];
  else if (c < 51) {
    int j = c - 3, band = j / 6, w = j % 6;
    float f = (float)(1 << band);
    v = (w < 3) ? sinf(coords[k * 3 + w] * f) : cosf(coords[k * 3 + (w - 3)] * f);
  }
  fco[idx] = (f16)v;
}

__global__ void hcombine_kernel(const float* __restrict__ xpart, const float* __restrict__ fcpart,
                                const float* __restrict__ node_b, f16* __restrict__ h16)
{
  long idx = (long)blockIdx.x * blockDim.x + threadIdx.x;
  if (idx >= NROWS * 512) return;
  int c = (int)(idx & 511);
  long bk = idx >> 9;
  int b = (int)(bk >> 12), k = (int)(bk & 4095);
  h16[idx] = (f16)(xpart[b * 512 + c] + fcpart[(long)k * 512 + c] + node_b[c]);
}

// ------------------------------ GAT / graph --------------------------------
__global__ void att_coef_kernel(const f16* __restrict__ xw, const float* __restrict__ wsrc,
                                const float* __restrict__ wdst, float* __restrict__ asrc,
                                float* __restrict__ adst, int NC, int OC)
{
  long idx = (long)blockIdx.x * blockDim.x + threadIdx.x;
  if (idx >= NROWS * 4) return;
  int hd = (int)(idx & 3);
  long bk = idx >> 2;
  const f16* row = xw + bk * NC + hd * OC;
  const float* ws = wsrc + hd * OC;
  const float* wd = wdst + hd * OC;
  float sa = 0.f, sd = 0.f;
  for (int c = 0; c < OC; ++c) {
    float v = (float)row[c];
    sa += v * ws[c]; sd += v * wd[c];
  }
  asrc[idx] = sa; adst[idx] = sd;
}

__global__ void edge_logits_kernel(const int* __restrict__ ei, const float* __restrict__ asrc,
                                   const float* __restrict__ adst, float* __restrict__ eb,
                                   float* __restrict__ seg)
{
  long idx = (long)blockIdx.x * blockDim.x + threadIdx.x;
  if (idx >= (long)BBATCH * NEL * 4) return;
  int hd = (int)(idx & 3);
  long t = idx >> 2;
  int i = (int)(t % NEL), b = (int)(t / NEL);
  int s, d;
  if (i < NEDGE) { s = ei[i]; d = ei[NEDGE + i]; } else { s = d = i - NEDGE; }
  float e = asrc[((long)b * KGR + s) * 4 + hd] + adst[((long)b * KGR + d) * 4 + hd];
  e = e > 0.f ? e : 0.2f * e;
  eb[idx] = e;
  atomicMaxF(&seg[((long)b * KGR + d) * 4 + hd], e);
}

__global__ void edge_exp_kernel(const int* __restrict__ ei, float* __restrict__ eb,
                                const float* __restrict__ seg, float* __restrict__ den)
{
  long idx = (long)blockIdx.x * blockDim.x + threadIdx.x;
  if (idx >= (long)BBATCH * NEL * 4) return;
  int hd = (int)(idx & 3);
  long t = idx >> 2;
  int i = (int)(t % NEL), b = (int)(t / NEL);
  int d = (i < NEDGE) ? ei[NEDGE + i] : i - NEDGE;
  float ex = __expf(eb[idx] - seg[((long)b * KGR + d) * 4 + hd]);
  eb[idx] = ex;
  atomicAdd(&den[((long)b * KGR + d) * 4 + hd], ex);
}

__global__ void edge_agg_kernel(const int* __restrict__ ei, const f16* __restrict__ xw,
                                const float* __restrict__ eb, const float* __restrict__ den,
                                float* __restrict__ out, int NC, int logOC)
{
  long t = blockIdx.x;
  int c = threadIdx.x;
  int i = (int)(t % NEL), b = (int)(t / NEL);
  int s, d;
  if (i < NEDGE) { s = ei[i]; d = ei[NEDGE + i]; } else { s = d = i - NEDGE; }
  int hd = c >> logOC;
  float alpha = eb[t * 4 + hd] / (den[((long)b * KGR + d) * 4 + hd] + 1e-16f);
  atomicAdd(&out[((long)b * KGR + d) * NC + c],
            (float)xw[((long)b * KGR + s) * NC + c] * alpha);
}

__global__ void bias_relu_f16_kernel(const float* __restrict__ in, const float* __restrict__ bias,
                                     f16* __restrict__ out, long n, int NC)
{
  long idx = (long)blockIdx.x * blockDim.x + threadIdx.x;
  if (idx >= n) return;
  out[idx] = (f16)fmaxf(in[idx] + bias[(int)(idx % NC)], 0.f);
}

__global__ void graph_agg_kernel(const int* __restrict__ ei, const f16* __restrict__ y,
                                 float* __restrict__ agg)
{
  long t = blockIdx.x;
  int c = threadIdx.x;
  int i = (int)(t % NEDGE), b = (int)(t / NEDGE);
  int s = ei[i], d = ei[NEDGE + i];
  atomicAdd(&agg[(((long)b * KGR + d) << 7) + c], (float)y[(((long)b * KGR + s) << 7) + c]);
}

__global__ void cvt32to16_kernel(const float* __restrict__ in, f16* __restrict__ out, long n)
{
  long idx = (long)blockIdx.x * blockDim.x + threadIdx.x;
  if (idx < n) out[idx] = (f16)in[idx];
}

__global__ void fill_f32_kernel(float* __restrict__ p, float v, long n)
{
  long idx = (long)blockIdx.x * blockDim.x + threadIdx.x;
  if (idx < n) p[idx] = v;
}

__global__ void head3_kernel(const f16* __restrict__ y4, const float* __restrict__ w,
                             const float* __restrict__ bias, float* __restrict__ out)
{
  long idx = (long)blockIdx.x * blockDim.x + threadIdx.x;
  if (idx >= NROWS * 3) return;
  long bk = idx / 3;
  int j = (int)(idx % 3);
  const f16* r = y4 + (bk << 7);
  float s = bias[j];
  for (int c = 0; c < 128; ++c) s += (float)r[c] * w[c * 3 + j];
  out[idx] = s;
}

// ---------------------------------------------------------------------------
extern "C" void kernel_launch(void* const* d_in, const int* in_sizes, int n_in,
                              void* d_out, int out_size, void* d_ws, size_t ws_size,
                              hipStream_t stream)
{
  // Input order = jax tree_leaves (alphabetical dict keys).
  const float* cage    = (const float*)d_in[0];
  const int*   eidx    = (const int*)d_in[1];
  const float* fc1_b   = (const float*)d_in[2];
  const float* fc1_w   = (const float*)d_in[3];
  const float* fc2_b   = (const float*)d_in[4];
  const float* fc2_w   = (const float*)d_in[5];
  const float* g1_W    = (const float*)d_in[6];
  const float* g1_adst = (const float*)d_in[7];
  const float* g1_asrc = (const float*)d_in[8];
  const float* g1_b    = (const float*)d_in[9];
  const float* g2_W    = (const float*)d_in[10];
  const float* g2_adst = (const float*)d_in[11];
  const float* g2_asrc = (const float*)d_in[12];
  const float* g2_b    = (const float*)d_in[13];
  const float* gc_b    = (const float*)d_in[14];
  const float* gc_wn   = (const float*)d_in[15];
  const float* gc_wr   = (const float*)d_in[16];
  const float* head_b  = (const float*)d_in[17];
  const float* head_w  = (const float*)d_in[18];
  const float* node_b  = (const float*)d_in[19];
  const float* node_w  = (const float*)d_in[20];
  const float* sensor  = (const float*)d_in[96];
  const float* t_norm  = (const float*)d_in[97];

  char* ws = (char*)d_ws;
  const size_t MB = 1ull << 20;
  // region 0 (64 MiB): H16, then reused as Y1 / XW2 / Y2
  f16*   H16    = (f16*)(ws + 0);
  f16*   Y1     = (f16*)(ws + 0);
  f16*   XW2    = (f16*)(ws + 32 * MB);
  f16*   Y2     = (f16*)(ws + 48 * MB);
  // region 1 (32 MiB): XW16, then reused as AGG16/Y4 + Y3
  f16*   XW16   = (f16*)(ws + 64 * MB);
  f16*   AGG16  = (f16*)(ws + 64 * MB);
  f16*   Y4     = (f16*)(ws + 64 * MB);
  f16*   Y3     = (f16*)(ws + 80 * MB);
  // region 2 (64 MiB): f32 atomic accumulators / GC accumulation / AGG32
  float* F32    = (float*)(ws + 96 * MB);
  float* GC32   = (float*)(ws + 96 * MB);
  float* AGG32  = (float*)(ws + 128 * MB);
  float* EB     = (float*)(ws + 160 * MB);      // 17 MiB
  float* SEG    = (float*)(ws + 178 * MB);
  float* DEN    = (float*)(ws + 179 * MB);
  float* ASRC   = (float*)(ws + 180 * MB);
  float* ADST   = (float*)(ws + 181 * MB);
  f16*   CNN0   = (f16*)(ws + 182 * MB);        // 16 MiB each
  f16*   CNN1   = (f16*)(ws + 198 * MB);
  const size_t SM = 214 * MB;
  float* POOL   = (float*)(ws + SM);
  float* FEAT   = (float*)(ws + SM + 0x10000);
  float* XPART  = (float*)(ws + SM + 0x20000);
  f16*   XCAT   = (f16*)(ws + SM + 0x30000);
  f16*   FCO    = (f16*)(ws + SM + 0x40000);
  float* FCPART = (float*)(ws + SM + 0xC0000);
  f16*   WNTOP  = (f16*)(ws + SM + 0x8C0000);
  f16*   WNBOT  = (f16*)(ws + SM + 0x950000);
  f16*   W1S    = (f16*)(ws + SM + 0x960000);
  f16*   W2S    = (f16*)(ws + SM + 0x9A0000);
  f16*   WROOT  = (f16*)(ws + SM + 0x9B0000);
  f16*   WNEI   = (f16*)(ws + SM + 0x9C0000);
  f16*   WFC1   = (f16*)(ws + SM + 0x9D0000);
  f16*   WCONV  = (f16*)(ws + SM + 0x9E0000);   // 14 * 294912 B
  float* BN     = (float*)(ws + SM + 0xDD0000); // 15 * (scale128+shift128)

  auto gemm = [&](const f16* A, const f16* Bf, const float* bias,
                  float* C32, f16* C16, int M, int N, int K, int addC, int relu) {
    dim3 g(N / 64, (M + 63) / 64), b(128);
    gemm_wmma_kernel<<<g, b, 8192, stream>>>(A, Bf, bias, C32, C16, M, N, K, addC, relu);
  };

  // ---- prep: BN folds + weight swizzles -----------------------------------
  bnfold_kernel<<<1, 128, 0, stream>>>((const float*)d_in[92], (const float*)d_in[91],
                                       (const float*)d_in[93], (const float*)d_in[94],
                                       BN, BN + 128);
  for (int r = 0; r < 7; ++r) {
    int base = 21 + r * 10;
    float* s1 = BN + (1 + 2 * r) * 256;
    float* s2 = BN + (2 + 2 * r) * 256;
    bnfold_kernel<<<1, 128, 0, stream>>>((const float*)d_in[base + 1], (const float*)d_in[base + 0],
                                         (const float*)d_in[base + 2], (const float*)d_in[base + 3],
                                         s1, s1 + 128);
    bnfold_kernel<<<1, 128, 0, stream>>>((const float*)d_in[base + 5], (const float*)d_in[base + 4],
                                         (const float*)d_in[base + 6], (const float*)d_in[base + 7],
                                         s2, s2 + 128);
    swizzle_conv_kernel<<<576, 256, 0, stream>>>((const float*)d_in[base + 8],
                                                 WCONV + (size_t)(2 * r) * 147456);
    swizzle_conv_kernel<<<576, 256, 0, stream>>>((const float*)d_in[base + 9],
                                                 WCONV + (size_t)(2 * r + 1) * 147456);
  }
  swizzle_w_kernel<<<1088, 256, 0, stream>>>(node_w, WNTOP, 544, 525, 512);
  swizzle_w_kernel<<<128, 256, 0, stream>>>(node_w + 525 * 512, WNBOT, 64, 51, 512);
  swizzle_w_kernel<<<512, 256, 0, stream>>>(g1_W, W1S, 512, 512, 256);
  swizzle_w_kernel<<<128, 256, 0, stream>>>(g2_W, W2S, 256, 256, 128);
  swizzle_w_kernel<<<64, 256, 0, stream>>>(gc_wr, WROOT, 128, 128, 128);
  swizzle_w_kernel<<<64, 256, 0, stream>>>(gc_wn, WNEI, 128, 128, 128);
  swizzle_w_kernel<<<64, 256, 0, stream>>>(fc1_w, WFC1, 128, 128, 128);

  // ---- CNN encoder --------------------------------------------------------
  stem_kernel<<<32768, 256, 0, stream>>>(sensor, (const float*)d_in[95], BN, BN + 128, CNN0);
  for (int r = 0; r < 7; ++r) {
    const f16* wc1 = WCONV + (size_t)(2 * r) * 147456;
    const f16* wc2 = WCONV + (size_t)(2 * r + 1) * 147456;
    float* s1 = BN + (1 + 2 * r) * 256;
    float* s2 = BN + (2 + 2 * r) * 256;
    conv3x3_wmma_kernel<<<dim3(2, 1024), 128, 0, stream>>>(CNN0, wc1, s1, s1 + 128, nullptr, CNN1);
    conv3x3_wmma_kernel<<<dim3(2, 1024), 128, 0, stream>>>(CNN1, wc2, s2, s2 + 128, CNN0, CNN0);
  }
  avgpool_kernel<<<2048, 256, 0, stream>>>(CNN0, POOL);
  head512_kernel<<<32, 256, 0, stream>>>(POOL, head_w, head_b, FEAT);

  // ---- node init (factored) ----------------------------------------------
  xcat_kernel<<<34, 256, 0, stream>>>(FEAT, t_norm, XCAT);
  fco_kernel<<<1024, 256, 0, stream>>>(cage, FCO);
  gemm(XCAT, WNTOP, nullptr, XPART, nullptr, 16, 512, 544, 0, 0);
  gemm(FCO, WNBOT, nullptr, FCPART, nullptr, 4096, 512, 64, 0, 0);
  hcombine_kernel<<<131072, 256, 0, stream>>>(XPART, FCPART, node_b, H16);

  // ---- GAT layer 1 (512 -> 256, 4 heads x 64) -----------------------------
  gemm(H16, W1S, nullptr, nullptr, XW16, 65536, 256, 512, 0, 0);
  att_coef_kernel<<<1024, 256, 0, stream>>>(XW16, g1_asrc, g1_adst, ASRC, ADST, 256, 64);
  fill_f32_kernel<<<1024, 256, 0, stream>>>(SEG, -1e30f, NROWS * 4);
  fill_f32_kernel<<<1024, 256, 0, stream>>>(DEN, 0.f, NROWS * 4);
  edge_logits_kernel<<<17408, 256, 0, stream>>>(eidx, ASRC, ADST, EB, SEG);
  edge_exp_kernel<<<17408, 256, 0, stream>>>(eidx, EB, SEG, DEN);
  fill_f32_kernel<<<65536, 256, 0, stream>>>(F32, 0.f, NROWS * 256);
  edge_agg_kernel<<<BBATCH * NEL, 256, 0, stream>>>(eidx, XW16, EB, DEN, F32, 256, 6);
  bias_relu_f16_kernel<<<65536, 256, 0, stream>>>(F32, g1_b, Y1, NROWS * 256, 256);

  // ---- GAT layer 2 (256 -> 128, 4 heads x 32) -----------------------------
  gemm(Y1, W2S, nullptr, nullptr, XW2, 65536, 128, 256, 0, 0);
  att_coef_kernel<<<1024, 256, 0, stream>>>(XW2, g2_asrc, g2_adst, ASRC, ADST, 128, 32);
  fill_f32_kernel<<<1024, 256, 0, stream>>>(SEG, -1e30f, NROWS * 4);
  fill_f32_kernel<<<1024, 256, 0, stream>>>(DEN, 0.f, NROWS * 4);
  edge_logits_kernel<<<17408, 256, 0, stream>>>(eidx, ASRC, ADST, EB, SEG);
  edge_exp_kernel<<<17408, 256, 0, stream>>>(eidx, EB, SEG, DEN);
  fill_f32_kernel<<<32768, 256, 0, stream>>>(F32, 0.f, NROWS * 128);
  edge_agg_kernel<<<BBATCH * NEL, 128, 0, stream>>>(eidx, XW2, EB, DEN, F32, 128, 5);
  bias_relu_f16_kernel<<<32768, 256, 0, stream>>>(F32, g2_b, Y2, NROWS * 128, 128);

  // ---- GraphConv + MLP head ----------------------------------------------
  fill_f32_kernel<<<32768, 256, 0, stream>>>(AGG32, 0.f, NROWS * 128);
  graph_agg_kernel<<<BBATCH * NEDGE, 128, 0, stream>>>(eidx, Y2, AGG32);
  cvt32to16_kernel<<<32768, 256, 0, stream>>>(AGG32, AGG16, NROWS * 128);
  gemm(Y2, WROOT, gc_b, GC32, nullptr, 65536, 128, 128, 0, 0);
  gemm(AGG16, WNEI, nullptr, GC32, Y3, 65536, 128, 128, 1, 1);
  gemm(Y3, WFC1, fc1_b, nullptr, Y4, 65536, 128, 128, 0, 1);
  head3_kernel<<<768, 256, 0, stream>>>(Y4, fc2_w, fc2_b, (float*)d_out);

  (void)in_sizes; (void)n_in; (void)out_size; (void)ws_size;
}
